// prototypeContrastLoss_7387343749233
// MI455X (gfx1250) — compile-verified, matrix operand order checked by reading, not run on hardware
//
#include <hip/hip_runtime.h>
#include <hip/hip_bf16.h>
#include <math.h>

typedef __attribute__((ext_vector_type(16))) __bf16 v16bf;
typedef __attribute__((ext_vector_type(8)))  float  v8f;
typedef __attribute__((address_space(3))) void lds_void_t;

#define NUM_B   16
#define DIM     512
#define HWPIX   4096          // 64*64
#define LABW    513
#define CPAD    32            // 21 classes padded to 32
#define NCLS    21
#define KSPLIT  8             // split HW across workgroups for occupancy

// ---------------------------------------------------------------------------
// 0) zero the atomic accumulators (counts + sums)
// ---------------------------------------------------------------------------
__global__ void init_ws(float* __restrict__ sums, int* __restrict__ counts) {
    int i = blockIdx.x * blockDim.x + threadIdx.x;
    if (i < NUM_B * CPAD) counts[i] = 0;
    if (i < NUM_B * CPAD * DIM) sums[i] = 0.0f;
}

// ---------------------------------------------------------------------------
// 1) bilinear label resize 513x513 -> 64x64 (half-pixel), trunc to int,
//    and per-(b,c) pixel counts
// ---------------------------------------------------------------------------
__global__ void resize_labels(const int* __restrict__ labels,
                              int* __restrict__ lab64,
                              int* __restrict__ counts) {
    int idx = blockIdx.x * blockDim.x + threadIdx.x;
    if (idx >= NUM_B * HWPIX) return;
    int b = idx >> 12;
    int hw = idx & (HWPIX - 1);
    int y = hw >> 6, x = hw & 63;

    const float scale = (float)LABW / 64.0f;   // 8.015625
    float fy = ((float)y + 0.5f) * scale - 0.5f;
    float fx = ((float)x + 0.5f) * scale - 0.5f;
    int y0 = (int)floorf(fy);  float wy = fy - (float)y0;
    int x0 = (int)floorf(fx);  float wx = fx - (float)x0;
    int y1 = min(y0 + 1, LABW - 1);  y0 = max(0, min(LABW - 1, y0));
    int x1 = min(x0 + 1, LABW - 1);  x0 = max(0, min(LABW - 1, x0));

    const int* lb = labels + (size_t)b * LABW * LABW;
    float v00 = (float)lb[y0 * LABW + x0];
    float v01 = (float)lb[y0 * LABW + x1];
    float v10 = (float)lb[y1 * LABW + x0];
    float v11 = (float)lb[y1 * LABW + x1];
    float val = v00 * (1.f - wy) * (1.f - wx) + v01 * (1.f - wy) * wx +
                v10 * wy * (1.f - wx)        + v11 * wy * wx;
    int cls = (int)val;                 // truncation, matches astype(int32)
    cls = max(0, min(NCLS - 1, cls));
    lab64[idx] = cls;
    atomicAdd(&counts[b * CPAD + cls], 1);
}

// ---------------------------------------------------------------------------
// 2) masked pooling as GEMM:  sums[b] += mask(CPADxHW) x features[b]^T (HWxD)
//    bf16 WMMA 16x16x32; feature staging via double-buffered
//    GLOBAL_LOAD_ASYNC_TO_LDS_B128 (ASYNCcnt) to overlap HBM with WMMA.
//    Labels for the whole 512-pixel K-slice hoisted to LDS once.
// ---------------------------------------------------------------------------
__global__ void __launch_bounds__(256)
pool_wmma(const float* __restrict__ feat, const int* __restrict__ lab64,
          float* __restrict__ sums) {
    int blk = blockIdx.x;
    int kc0 = (blk % KSPLIT) * (HWPIX / KSPLIT);  // 512-pixel K slice
    int t   = blk / KSPLIT;
    int d0  = (t & 7) * 64;                       // 8 d-tiles of 64
    int b   = t >> 3;

    int tid  = threadIdx.x;
    int lane = tid & 31;
    int wave = tid >> 5;
    int mtile = wave & 1;                         // classes 0..15 / 16..31
    int ntile = wave >> 1;                        // d sub-tile 0..3
    int lan   = lane & 15;
    int laneHi = lane >> 4;

    // row stride 36 floats = 144 B: multiple of 16 B (B128-aligned) and
    // 36*n mod 64 distinct for n=0..15 -> conflict-free fragment gather
    __shared__ __attribute__((aligned(16))) float featLds[2][64][36];
    __shared__ int labLds[HWPIX / KSPLIT];        // 512 labels, staged once

    const float* fbase = feat + (size_t)b * DIM * HWPIX;
    const int*   lbase = lab64 + b * HWPIX + kc0;

    labLds[tid]       = lbase[tid];
    labLds[256 + tid] = lbase[256 + tid];

    int dl = tid >> 2;                            // 0..63 : local d row
    int kg = (tid & 3) * 8;                       // 0,8,16,24 : k sub-group

    // async-stage one 64x32 f32 chunk into featLds[bufIdx]
    auto stage = [&](int bufIdx, int kc) {
        unsigned ldsA = (unsigned)(size_t)(lds_void_t*)&featLds[bufIdx][dl][kg];
        unsigned goff = (((unsigned)(d0 + dl) << 12) +
                         (unsigned)(kc0 + kc + kg)) << 2;   // byte offset
        asm volatile(
            "global_load_async_to_lds_b128 %0, %1, %2\n\t"
            "global_load_async_to_lds_b128 %0, %1, %2 offset:16"
            :: "v"(ldsA), "v"(goff), "s"(fbase) : "memory");
    };

    stage(0, 0);
    asm volatile("s_wait_asynccnt 0" ::: "memory");
    __syncthreads();                              // also covers labLds

    v8f acc = {};
    const __bf16 one  = (__bf16)1.0f;
    const __bf16 zero = (__bf16)0.0f;
    const int nIter = (HWPIX / KSPLIT) / 32;      // 16

    for (int it = 0; it < nIter; ++it) {
        int p = it & 1;
        if (it + 1 < nIter) stage(1 - p, (it + 1) * 32);   // prefetch in flight

        int kcl = it * 32;
        v16bf afrag, bfrag;
        int mrow = mtile * 16 + lan;
        int ncol = ntile * 16 + lan;
#pragma unroll
        for (int v = 0; v < 8; ++v) {
            int kb = ((v & 4) ? 16 : 0) + laneHi * 8 + (v & 3) * 2;
            int c0 = labLds[kcl + kb], c1 = labLds[kcl + kb + 1];
            afrag[2 * v]     = (c0 == mrow) ? one : zero;
            afrag[2 * v + 1] = (c1 == mrow) ? one : zero;
            bfrag[2 * v]     = (__bf16)featLds[p][ncol][kb];
            bfrag[2 * v + 1] = (__bf16)featLds[p][ncol][kb + 1];
        }
        acc = __builtin_amdgcn_wmma_f32_16x16x32_bf16(
                  false, afrag, false, bfrag, (short)0, acc, false, false);

        if (it + 1 < nIter) {
            asm volatile("s_wait_asynccnt 0" ::: "memory");
            __syncthreads();
        }
    }

    int n  = d0 + ntile * 16 + lan;
    int mb = mtile * 16 + laneHi * 8;
#pragma unroll
    for (int r = 0; r < 8; ++r) {
        atomicAdd(&sums[((size_t)b * CPAD + mb + r) * DIM + n], acc[r]);
    }
}

// ---------------------------------------------------------------------------
// 3) L2-normalize prototypes, emit bf16 rows [512 x 512]
// ---------------------------------------------------------------------------
__global__ void __launch_bounds__(256)
normalize_protos(const float* __restrict__ sums, __bf16* __restrict__ protos) {
    __shared__ float red[8];
    __shared__ float invS;
    int row = blockIdx.x;                 // b*32 + c  (512 rows)
    int tid = threadIdx.x;
    float v0 = sums[(size_t)row * DIM + tid];
    float v1 = sums[(size_t)row * DIM + 256 + tid];
    float ss = v0 * v0 + v1 * v1;
#pragma unroll
    for (int off = 16; off; off >>= 1) ss += __shfl_down(ss, off, 32);
    if ((tid & 31) == 0) red[tid >> 5] = ss;
    __syncthreads();
    if (tid == 0) {
        float tot = 0.f;
#pragma unroll
        for (int i = 0; i < 8; ++i) tot += red[i];
        invS = 1.0f / fmaxf(sqrtf(tot), 1e-12f);
    }
    __syncthreads();
    float inv = invS;
    protos[(size_t)row * DIM + tid]       = (__bf16)(v0 * inv);
    protos[(size_t)row * DIM + 256 + tid] = (__bf16)(v1 * inv);
}

// ---------------------------------------------------------------------------
// 4) Gram matrix G = P * P^T  (512x512 f32) with bf16 WMMA
// ---------------------------------------------------------------------------
__global__ void __launch_bounds__(256)
gram_wmma(const __bf16* __restrict__ protos, float* __restrict__ G) {
    int m0 = (blockIdx.x >> 3) * 32;      // 16 M-blocks of 32 rows
    int n0 = (blockIdx.x & 7) * 64;       // 8 N-blocks of 64 cols
    int tid = threadIdx.x, lane = tid & 31, wave = tid >> 5;
    int mtile = wave & 1, ntile = wave >> 1;
    int lan = lane & 15, laneHi = lane >> 4;

    __shared__ __bf16 aLds[32][34];
    __shared__ __bf16 bLds[64][34];
    v8f acc = {};

    for (int kc = 0; kc < DIM; kc += 32) {
        {   // A tile: 32 rows x 32 k, 4 bf16 per thread
            int r = tid >> 3, k = (tid & 7) * 4;
            const __bf16* src = protos + (size_t)(m0 + r) * DIM + kc + k;
#pragma unroll
            for (int u = 0; u < 4; ++u) aLds[r][k + u] = src[u];
        }
        {   // B tile (cols of G = rows of P): 64 rows x 32 k, 8 bf16/thread
            int r = tid >> 2, k = (tid & 3) * 8;
            const __bf16* src = protos + (size_t)(n0 + r) * DIM + kc + k;
#pragma unroll
            for (int u = 0; u < 8; ++u) bLds[r][k + u] = src[u];
        }
        __syncthreads();

        v16bf afrag, bfrag;
#pragma unroll
        for (int v = 0; v < 8; ++v) {
            int kb = ((v & 4) ? 16 : 0) + laneHi * 8 + (v & 3) * 2;
            afrag[2 * v]     = aLds[mtile * 16 + lan][kb];
            afrag[2 * v + 1] = aLds[mtile * 16 + lan][kb + 1];
            bfrag[2 * v]     = bLds[ntile * 16 + lan][kb];
            bfrag[2 * v + 1] = bLds[ntile * 16 + lan][kb + 1];
        }
        acc = __builtin_amdgcn_wmma_f32_16x16x32_bf16(
                  false, afrag, false, bfrag, (short)0, acc, false, false);
        __syncthreads();
    }

    int n  = n0 + ntile * 16 + lan;
    int mb = m0 + mtile * 16 + laneHi * 8;
#pragma unroll
    for (int r = 0; r < 8; ++r) {
        G[(size_t)(mb + r) * DIM + n] = acc[r];
    }
}

// ---------------------------------------------------------------------------
// 5) masked logsumexp epilogue: one wave per class (21 waves)
// ---------------------------------------------------------------------------
__global__ void __launch_bounds__(672)
loss_kernel(const float* __restrict__ G, const int* __restrict__ counts,
            float* __restrict__ out) {
    __shared__ float contribS[NCLS];
    int c    = threadIdx.x >> 5;          // class = wave id, 0..20
    int lane = threadIdx.x & 31;
    const float NEG  = -1e9f;
    const float invT = 10.0f;             // 1 / TEMPERATURE

    // ---- all-blocks logsumexp over (e,i,j): 21*16*16 = 5376 entries ----
    float mAll = -INFINITY;
    for (int idx = lane; idx < NCLS * 256; idx += 32) {
        int e = idx >> 8, i = (idx >> 4) & 15, j = idx & 15;
        bool v = (counts[i * CPAD + c] > 0) && (counts[j * CPAD + e] > 0);
        float s = v ? G[(size_t)(i * CPAD + c) * DIM + (j * CPAD + e)] * invT : NEG;
        mAll = fmaxf(mAll, s);
    }
#pragma unroll
    for (int off = 16; off; off >>= 1) mAll = fmaxf(mAll, __shfl_xor(mAll, off, 32));
    float sAll = 0.f;
    for (int idx = lane; idx < NCLS * 256; idx += 32) {
        int e = idx >> 8, i = (idx >> 4) & 15, j = idx & 15;
        bool v = (counts[i * CPAD + c] > 0) && (counts[j * CPAD + e] > 0);
        float s = v ? G[(size_t)(i * CPAD + c) * DIM + (j * CPAD + e)] * invT : NEG;
        sAll += __expf(s - mAll);
    }
#pragma unroll
    for (int off = 16; off; off >>= 1) sAll += __shfl_xor(sAll, off, 32);

    // ---- diagonal block (positives): 16*16 entries ----
    float mD = -INFINITY;
    for (int idx = lane; idx < 256; idx += 32) {
        int i = idx >> 4, j = idx & 15;
        bool v = (counts[i * CPAD + c] > 0) && (counts[j * CPAD + c] > 0);
        float s = v ? G[(size_t)(i * CPAD + c) * DIM + (j * CPAD + c)] * invT : NEG;
        mD = fmaxf(mD, s);
    }
#pragma unroll
    for (int off = 16; off; off >>= 1) mD = fmaxf(mD, __shfl_xor(mD, off, 32));
    float sD = 0.f, dsum = 0.f;
    int cnt = 0;
    for (int idx = lane; idx < 256; idx += 32) {
        int i = idx >> 4, j = idx & 15;
        bool v = (counts[i * CPAD + c] > 0) && (counts[j * CPAD + c] > 0);
        float s = v ? G[(size_t)(i * CPAD + c) * DIM + (j * CPAD + c)] * invT : NEG;
        sD += __expf(s - mD);
        dsum += v ? s : 0.f;
        cnt  += v ? 1 : 0;
    }
#pragma unroll
    for (int off = 16; off; off >>= 1) {
        sD   += __shfl_xor(sD, off, 32);
        dsum += __shfl_xor(dsum, off, 32);
        cnt  += __shfl_xor(cnt, off, 32);
    }

    if (lane == 0) {
        float lseAll = mAll + __logf(sAll);
        float lseD   = mD + __logf(sD);
        float mm  = fmaxf(lseAll, lseD);
        float lse = mm + __logf(__expf(lseAll - mm) + __expf(lseD - mm));
        float pos_mean = dsum / (float)max(cnt, 1);
        bool present = false;
        for (int b = 0; b < NUM_B; ++b) present |= (counts[b * CPAD + c] > 0);
        contribS[c] = present ? (lse - pos_mean) : 0.f;
    }
    __syncthreads();
    if (threadIdx.x == 0) {
        float loss = 0.f;
        for (int k = 0; k < NCLS; ++k) loss += contribS[k];
        out[0] = loss;
    }
}

// ---------------------------------------------------------------------------
extern "C" void kernel_launch(void* const* d_in, const int* in_sizes, int n_in,
                              void* d_out, int out_size, void* d_ws, size_t ws_size,
                              hipStream_t stream) {
    (void)in_sizes; (void)n_in; (void)out_size; (void)ws_size;
    const float* feat   = (const float*)d_in[0];   // [16,512,64,64]
    const int*   labels = (const int*)d_in[1];     // [16,513,513]
    float* out = (float*)d_out;

    char* ws = (char*)d_ws;
    int*    lab64  = (int*)(ws);                               // 256 KiB
    int*    counts = (int*)(ws + 262144);                      // 2 KiB
    float*  sums   = (float*)(ws + 266240);                    // 1 MiB
    __bf16* protos = (__bf16*)(ws + 266240 + 1048576);         // 512 KiB
    float*  G      = (float*)(ws + 266240 + 1048576 + 524288); // 1 MiB

    // 0) zero accumulators
    init_ws<<<(NUM_B * CPAD * DIM + 255) / 256, 256, 0, stream>>>(sums, counts);
    // 1) label resize + counts
    resize_labels<<<(NUM_B * HWPIX + 255) / 256, 256, 0, stream>>>(labels, lab64, counts);
    // 2) masked pooling GEMM (HBM-bound: 128 MiB -> ~5.5us roofline)
    pool_wmma<<<NUM_B * (DIM / 64) * KSPLIT, 256, 0, stream>>>(feat, lab64, sums);
    // 3) normalize prototypes -> bf16
    normalize_protos<<<NUM_B * CPAD, 256, 0, stream>>>(sums, protos);
    // 4) Gram matrix with WMMA
    gram_wmma<<<(DIM / 32) * (DIM / 64), 256, 0, stream>>>(protos, G);
    // 5) masked logsumexp -> scalar loss
    loss_kernel<<<1, NCLS * 32, 0, stream>>>(G, counts, out);
}
